// GemmaLinearAttention_43808666419801
// MI455X (gfx1250) — matrix operand order repaired and checked by reading, not compiled, not run on hardware
//
#include <hip/hip_runtime.h>

typedef __bf16 bf16_t;
typedef __attribute__((ext_vector_type(16))) __bf16 v16bf;
typedef __attribute__((ext_vector_type(8)))  __bf16 v8bf;
typedef __attribute__((ext_vector_type(8)))  float  v8f;
typedef __attribute__((ext_vector_type(4)))  int    v4i;
typedef __attribute__((ext_vector_type(8)))  int    v8i;

#define WMMA_BF16(a,b,c) __builtin_amdgcn_wmma_f32_16x16x32_bf16(false,(a),false,(b),(short)0,(c),false,false)

// ---- CDNA5 async global->LDS copy (ASYNCcnt) ----
__device__ __forceinline__ void async_load_b128(const void* gptr, void* lptr) {
  unsigned long long g = (unsigned long long)gptr;
  unsigned l = (unsigned)(unsigned long long)lptr;   // generic->LDS: low 32 bits
  asm volatile("global_load_async_to_lds_b128 %0, %1, off" :: "v"(l), "v"(g) : "memory");
}
#define WAIT_ASYNC(n) asm volatile("s_wait_asynccnt " #n ::: "memory")

// ---- CDNA5 Tensor Data Mover: 2D tile (128 rows x 64 bf16) -> LDS with hw padding ----
// D# per CDNA5 ISA. Pads 4 DWORDs every 32 DWORDs: LDS rows of 72 bf16.
__device__ __forceinline__ void tdm_load_tile_b(const bf16_t* gsrc, unsigned lds_byte,
                                                int tensorN, int tensorK) {
  unsigned long long ga = (unsigned long long)gsrc;
  v4i g0;
  g0[0] = 1;                                            // count=1, user descriptor
  g0[1] = (int)lds_byte;                                // lds_addr (bytes)
  g0[2] = (int)(ga & 0xFFFFFFFFull);                    // global_addr[31:0]
  g0[3] = (int)(((ga >> 32) & 0x01FFFFFFull) | (2u << 30)); // global_addr[56:32] | type=2
  v8i g1;
  g1[0] = (1 << 16)      // data_size = 2 bytes
        | (1 << 20)      // pad_enable
        | (4 << 22)      // pad_interval: 32 DWORDs (= 64 bf16 per row)
        | (3 << 25);     // pad_amount: 4 DWORDs (= 8 bf16 pad -> LDS row of 72)
  g1[1] = (int)(((unsigned)tensorK & 0xFFFFu) << 16);               // tensor_dim0[15:0]
  g1[2] = (int)(((unsigned)tensorK >> 16) | (((unsigned)tensorN & 0xFFFFu) << 16)); // dim0 hi | dim1 lo
  g1[3] = (int)(((unsigned)tensorN >> 16) | (64u << 16));           // dim1 hi | tile_dim0=64
  g1[4] = 128;                                                      // tile_dim1=128, tile_dim2=0
  g1[5] = tensorK;                                                  // tensor_dim0_stride[31:0]
  g1[6] = 0;                                                        // stride0 hi | stride1 lo
  g1[7] = 0;
  asm volatile("tensor_load_to_lds %0, %1" :: "s"(g0), "s"(g1) : "memory");
}

// ---------------- WMMA fragment loaders (bf16, 16x16x32) ----------------
// A matrix 16x32: lane(0..15)=row M, VGPR0..3 = K(g*8+0..7), VGPR4..7 = K(16+g*8+0..7)
__device__ __forceinline__ v16bf ld_frag_a(const bf16_t* p0, int ld) {
  int lane = threadIdx.x & 31;
  int r = lane & 15, g = lane >> 4;
  const bf16_t* p = p0 + r * ld + g * 8;
  v8bf lo = *(const v8bf*)(p);
  v8bf hi = *(const v8bf*)(p + 16);
  v16bf o;
#pragma unroll
  for (int i = 0; i < 8; ++i) { o[i] = lo[i]; o[i + 8] = hi[i]; }
  return o;
}

// B matrix 32x16 stored transposed in LDS as [N][K]: lane(0..15)=col N, VGPR0..3 = K(g*8+0..7), VGPR4..7 = K(16+g*8+0..7)
__device__ __forceinline__ v16bf ld_frag_b(const bf16_t* p0, int ld) {
  int lane = threadIdx.x & 31;
  int nn = lane & 15, g = lane >> 4;
  const bf16_t* p = p0 + nn * ld + g * 8;
  v8bf lo = *(const v8bf*)(p);
  v8bf hi = *(const v8bf*)(p + 16);
  v16bf o;
#pragma unroll
  for (int i = 0; i < 8; ++i) { o[i] = lo[i]; o[i + 8] = hi[i]; }
  return o;
}

// ---------------- elementwise conversion kernels ----------------
__global__ __launch_bounds__(256) void cvt_f32_bf16(const float* __restrict__ in,
                                                    bf16_t* __restrict__ out, int n) {
  int i = blockIdx.x * 256 + threadIdx.x;
  if (i < n) out[i] = (bf16_t)in[i];
}

// in [R][C] f32 -> out [C][R] bf16
__global__ __launch_bounds__(256) void transpose_cvt(const float* __restrict__ in,
                                                     bf16_t* __restrict__ out, int R, int C) {
  int i = blockIdx.x * 256 + threadIdx.x;
  if (i >= R * C) return;
  int r = i / C, c = i % C;
  out[(size_t)c * R + r] = (bf16_t)in[(size_t)r * C + c];
}

// fm [H=8][256][64] f32 -> fmT [8][64][256] bf16
__global__ __launch_bounds__(256) void fm_transpose(const float* __restrict__ in,
                                                    bf16_t* __restrict__ out) {
  int i = blockIdx.x * 256 + threadIdx.x;
  if (i >= 8 * 256 * 64) return;
  int f = i % 64, d = (i / 64) % 256, h = i / (64 * 256);
  out[((size_t)h * 64 + f) * 256 + d] = (bf16_t)in[i];
}

// ---------------- tiled WMMA GEMM: async A-tile + TDM B-tile, double buffered ----------------
#define GEMM_BK 64
#define GEMM_LDA (GEMM_BK + 8)   // 72
template <int N, int K>
__global__ __launch_bounds__(256) void gemm_bf16(const bf16_t* __restrict__ A,
                                                 const bf16_t* __restrict__ BT,
                                                 float* __restrict__ C) {
  extern __shared__ char smem[];
  bf16_t* sA[2]; bf16_t* sB[2];
  sA[0] = (bf16_t*)smem;                 // [128][72]
  sB[0] = sA[0] + 128 * GEMM_LDA;
  sA[1] = sB[0] + 128 * GEMM_LDA;
  sB[1] = sA[1] + 128 * GEMM_LDA;
  int tid = threadIdx.x, lane = tid & 31, wave = tid >> 5;
  int waveM = wave >> 2;  // 0..1 -> 64 rows
  int waveN = wave & 3;   // 0..3 -> 32 cols
  int bm = blockIdx.y * 128;
  int bn = blockIdx.x * 128;

  // A tile: per-wave async fills (4 x b128 per thread)
  auto issue_A = [&](int k0, int buf) {
#pragma unroll
    for (int i = 0; i < 4; ++i) {
      int t = tid + i * 256;        // 0..1023
      int r = t >> 3;               // 0..127
      int c = (t & 7) * 8;          // 0..56
      async_load_b128(A + (size_t)(bm + r) * K + k0 + c, sA[buf] + r * GEMM_LDA + c);
    }
  };
  // B tile: one TDM descriptor issued by wave 0 (EXEC-independent, TENSORcnt tracked)
  auto issue_B = [&](int k0, int buf) {
    if (wave == 0)
      tdm_load_tile_b(BT + (size_t)bn * K + k0, (unsigned)(unsigned long long)sB[buf], N, K);
  };

  v8f acc[4][2] = {};
  const int NK = K / GEMM_BK;
  issue_A(0, 0);
  issue_B(0, 0);
  for (int kt = 0; kt < NK; ++kt) {
    int buf = kt & 1;
    if (kt + 1 < NK) {
      issue_A((kt + 1) * GEMM_BK, buf ^ 1);
      issue_B((kt + 1) * GEMM_BK, buf ^ 1);
      if ((kt + 8) * GEMM_BK < K) {      // L2 prefetch far ahead (A stream)
        int r = tid >> 3, c = (tid & 7) * 8;
        __builtin_prefetch(A + (size_t)(bm + r) * K + (kt + 8) * GEMM_BK + c, 0, 1);
      }
      WAIT_ASYNC(4);                               // current A tile resident; next in flight
      if (wave == 0) __builtin_amdgcn_s_wait_tensorcnt(1);  // current B tile resident
    } else {
      WAIT_ASYNC(0);
      if (wave == 0) __builtin_amdgcn_s_wait_tensorcnt(0);
    }
    __syncthreads();      // publish all fills of this buffer
#pragma unroll
    for (int kk = 0; kk < 2; ++kk) {
      v16bf af[4], bfr[2];
#pragma unroll
      for (int i = 0; i < 4; ++i)
        af[i] = ld_frag_a(sA[buf] + (waveM * 64 + i * 16) * GEMM_LDA + kk * 32, GEMM_LDA);
#pragma unroll
      for (int j = 0; j < 2; ++j)
        bfr[j] = ld_frag_b(sB[buf] + (waveN * 32 + j * 16) * GEMM_LDA + kk * 32, GEMM_LDA);
#pragma unroll
      for (int i = 0; i < 4; ++i)
#pragma unroll
        for (int j = 0; j < 2; ++j)
          acc[i][j] = WMMA_BF16(af[i], bfr[j], acc[i][j]);
    }
    __syncthreads();      // done reading buffer before it is refilled
  }
  int g = lane >> 4, cn = lane & 15;
#pragma unroll
  for (int i = 0; i < 4; ++i)
#pragma unroll
    for (int j = 0; j < 2; ++j) {
      int row0 = bm + waveM * 64 + i * 16 + 8 * g;
      int col = bn + waveN * 32 + j * 16 + cn;
      float* cp = C + (size_t)row0 * N + col;
#pragma unroll
      for (int r = 0; r < 8; ++r)
        cp[(size_t)r * N] = acc[i][j][r];
    }
}

// ---------------- RoPE: Xin[B*L][XH*256] f32 -> Xout[B][XH][L][256] bf16 ----------------
__global__ __launch_bounds__(256) void rope_kernel(const float* __restrict__ Xin,
                                                   const float* __restrict__ Cos,
                                                   const float* __restrict__ Sin,
                                                   bf16_t* __restrict__ Xout, int XH) {
  int idx = blockIdx.x * 256 + threadIdx.x;
  int total = 2 * 2048 * XH * 128;
  if (idx >= total) return;
  int d = idx & 127;
  int l = (idx >> 7) & 2047;
  int rest = idx >> 18;
  int h = rest % XH;
  int b = rest / XH;
  const float* row = Xin + (size_t)(b * 2048 + l) * (XH * 256) + h * 256;
  float x1 = row[d], x2 = row[d + 128];
  float c = Cos[l * 128 + d], s = Sin[l * 128 + d];
  bf16_t* orow = Xout + (((size_t)(b * XH + h) * 2048) + l) * 256;
  orow[d]       = (bf16_t)(x1 * c - x2 * s);
  orow[d + 128] = (bf16_t)(x1 * s + x2 * c);
}

// ---------------- hedgehog: P = X[64,256] @ fmT[h]^T -> softmax(concat(P,-P)) ----------------
__global__ __launch_bounds__(256) void hedgehog_kernel(const bf16_t* __restrict__ X,    // [B][XH][L][256]
                                                       const bf16_t* __restrict__ FmT,  // [8][64][256]
                                                       bf16_t* __restrict__ Phi,        // [B][8][L][128]
                                                       int XH, float scale) {
  extern __shared__ char smem[];
  const int LDX = 256 + 8;   // 264
  const int LDP = 64 + 8;    // 72
  bf16_t* sX = (bf16_t*)smem;            // [64][264]
  bf16_t* sF = sX + 64 * LDX;            // [64][264]
  float*  sP = (float*)(sF + 64 * LDX);  // [64][72]
  int tid = threadIdx.x, lane = tid & 31, wave = tid >> 5;
  int lt = blockIdx.x;
  int bh = blockIdx.y;
  int b = bh >> 3, h = bh & 7;
  const bf16_t* xbase = X + ((size_t)(b * XH + (XH == 1 ? 0 : h)) * 2048 + (size_t)lt * 64) * 256;
  const bf16_t* fbase = FmT + (size_t)h * 64 * 256;
#pragma unroll
  for (int i = 0; i < 8; ++i) {
    int t = tid + i * 256;       // 0..2047
    int r = t >> 5;              // 0..63
    int c = (t & 31) * 8;        // 0..248
    async_load_b128(xbase + (size_t)r * 256 + c, sX + r * LDX + c);
    async_load_b128(fbase + (size_t)r * 256 + c, sF + r * LDX + c);
  }
  WAIT_ASYNC(0);
  __syncthreads();
  int mi = wave >> 1, njb = (wave & 1) * 2;
  v8f acc[2] = {};
  for (int kk = 0; kk < 8; ++kk) {
    v16bf a = ld_frag_a(sX + (mi * 16) * LDX + kk * 32, LDX);
#pragma unroll
    for (int j = 0; j < 2; ++j) {
      v16bf bb = ld_frag_b(sF + ((njb + j) * 16) * LDX + kk * 32, LDX);
      acc[j] = WMMA_BF16(a, bb, acc[j]);
    }
  }
  int g = lane >> 4, cn = lane & 15;
#pragma unroll
  for (int j = 0; j < 2; ++j)
#pragma unroll
    for (int r = 0; r < 8; ++r)
      sP[(mi * 16 + r + 8 * g) * LDP + (njb + j) * 16 + cn] = acc[j][r];
  __syncthreads();
  if (tid < 64) {
    const float* prow = sP + tid * LDP;
    float m = 0.f;
    for (int j = 0; j < 64; ++j) m = fmaxf(m, fabsf(prow[j]));
    float sum = 0.f;
    for (int j = 0; j < 64; ++j) {
      float pv = prow[j];
      sum += __expf(pv - m) + __expf(-pv - m);
    }
    float inv = scale / sum;
    bf16_t* orow = Phi + (((size_t)(b * 8 + h) * 2048) + (size_t)lt * 64 + tid) * 128;
    for (int j = 0; j < 64; ++j) {
      float pv = prow[j];
      orow[j]      = (bf16_t)(__expf(pv - m) * inv);
      orow[j + 64] = (bf16_t)(__expf(-pv - m) * inv);
    }
  }
}

// ---------------- chunked linear attention ----------------
// grid = (16 b*h, 4 dv-slices); KV state (128 x 64-slice) kept as WMMA accumulators
__global__ __launch_bounds__(256) void chunk_attn_kernel(const bf16_t* __restrict__ PhiQ, // [B][8][L][128]
                                                         const bf16_t* __restrict__ PhiK, // [B][8][L][128]
                                                         const float*  __restrict__ V,    // [B*L][256]
                                                         bf16_t* __restrict__ Out) {      // [B*L][2048]
  extern __shared__ char smem[];
  const int LDF = 128 + 8;   // 136
  const int LDC = 64 + 8;    // 72
  bf16_t* sQ  = (bf16_t*)smem;           // [64][136]  q chunk (A layout)
  bf16_t* sK  = sQ  + 64 * LDF;          // [64][136]  k chunk (B layout for s = q k^T)
  bf16_t* sKT = sK  + 64 * LDF;          // [128][72]  k^T (A layout for kv update)
  bf16_t* sVT = sKT + 128 * LDC;         // [64][72]   v^T (B layout)
  bf16_t* sS  = sVT + 64 * LDC;          // [64][72]   masked scores (A layout)
  bf16_t* sKV = sS  + 64 * LDC;          // [64][136]  KV state dumped as [dv][f] (B layout)
  int tid = threadIdx.x, lane = tid & 31, wave = tid >> 5;
  int bh = blockIdx.x;            // 0..15
  int dvs = blockIdx.y;           // 0..3
  int b = bh >> 3, h = bh & 7;
  const bf16_t* qbase = PhiQ + (size_t)bh * 2048 * 128;
  const bf16_t* kbase = PhiK + (size_t)bh * 2048 * 128;
  const float*  vbase = V + (size_t)b * 2048 * 256 + dvs * 64;
  int g = lane >> 4, cn = lane & 15;
  int waveM = wave >> 1, waveN = wave & 1;

  v8f kv[4] = {};   // KV state: f-tile = wave (16 rows), dv-tiles 0..3

  for (int n = 0; n < 32; ++n) {
    __syncthreads();
    // dump pre-update KV state to LDS as bf16 [dv][f]
#pragma unroll
    for (int j = 0; j < 4; ++j)
#pragma unroll
      for (int r = 0; r < 8; ++r)
        sKV[(j * 16 + cn) * LDF + wave * 16 + r + 8 * g] = (bf16_t)kv[j][r];
    // async-fill q and k chunks (raw copies)
#pragma unroll
    for (int i = 0; i < 4; ++i) {
      int t = tid + i * 256;      // 0..1023
      int r = t >> 4;             // chunk row 0..63
      int c = (t & 15) * 8;       // f 0..120
      async_load_b128(qbase + (size_t)(n * 64 + r) * 128 + c, sQ + r * LDF + c);
      async_load_b128(kbase + (size_t)(n * 64 + r) * 128 + c, sK + r * LDF + c);
    }
    // v^T slice (f32 -> bf16 conversion, overlaps with async fill)
#pragma unroll
    for (int i = 0; i < 16; ++i) {
      int t = tid + i * 256;      // 0..4095
      int c = t >> 6;             // chunk row
      int d = t & 63;             // dv
      sVT[d * LDC + c] = (bf16_t)vbase[(size_t)(n * 64 + c) * 256 + d];
    }
    WAIT_ASYNC(0);
    __syncthreads();              // sQ, sK, sVT, sKV ready
    // build k^T from LDS
#pragma unroll
    for (int i = 0; i < 4; ++i) {
      int t = tid + i * 256;
      int r = t >> 4;
      int c = (t & 15) * 8;
      v8bf kk = *(const v8bf*)(sK + r * LDF + c);
#pragma unroll
      for (int e = 0; e < 8; ++e) sKT[(c + e) * LDC + r] = kk[e];
    }
    // inter: out = q @ KVstate
    v8f oacc[2] = {};
#pragma unroll
    for (int j = 0; j < 2; ++j) {
      int nt = waveN * 2 + j;
      for (int kk = 0; kk < 4; ++kk) {
        v16bf a  = ld_frag_a(sQ  + (waveM * 16) * LDF + kk * 32, LDF);
        v16bf bb = ld_frag_b(sKV + (nt * 16) * LDF + kk * 32, LDF);
        oacc[j] = WMMA_BF16(a, bb, oacc[j]);
      }
    }
    // s = tril(q @ k^T)
#pragma unroll
    for (int j = 0; j < 2; ++j) {
      int nt = waveN * 2 + j;
      v8f sacc = {};
      for (int kk = 0; kk < 4; ++kk) {
        v16bf a  = ld_frag_a(sQ + (waveM * 16) * LDF + kk * 32, LDF);
        v16bf bb = ld_frag_b(sK + (nt * 16) * LDF + kk * 32, LDF);
        sacc = WMMA_BF16(a, bb, sacc);
      }
#pragma unroll
      for (int r = 0; r < 8; ++r) {
        int ii = waveM * 16 + r + 8 * g;
        int jj = nt * 16 + cn;
        sS[ii * LDC + jj] = (bf16_t)((jj <= ii) ? sacc[r] : 0.f);
      }
    }
    __syncthreads();              // sKT and sS ready
    // KV state update: kv += k^T @ v  (inter already consumed old state from sKV)
#pragma unroll
    for (int j = 0; j < 4; ++j)
#pragma unroll
      for (int kk = 0; kk < 2; ++kk) {
        v16bf a  = ld_frag_a(sKT + (wave * 16) * LDC + kk * 32, LDC);
        v16bf bb = ld_frag_b(sVT + (j * 16) * LDC + kk * 32, LDC);
        kv[j] = WMMA_BF16(a, bb, kv[j]);
      }
    // intra: out += s @ v
#pragma unroll
    for (int j = 0; j < 2; ++j) {
      int nt = waveN * 2 + j;
#pragma unroll
      for (int kk = 0; kk < 2; ++kk) {
        v16bf a  = ld_frag_a(sS + (waveM * 16) * LDC + kk * 32, LDC);
        v16bf bb = ld_frag_b(sVT + (nt * 16) * LDC + kk * 32, LDC);
        oacc[j] = WMMA_BF16(a, bb, oacc[j]);
      }
    }
    // store output tile (bf16, layout [B*L][H*HD] for final GEMM)
#pragma unroll
    for (int j = 0; j < 2; ++j) {
      int nt = waveN * 2 + j;
#pragma unroll
      for (int r = 0; r < 8; ++r) {
        int row = b * 2048 + n * 64 + waveM * 16 + r + 8 * g;
        int col = h * 256 + dvs * 64 + nt * 16 + cn;
        Out[(size_t)row * 2048 + col] = (bf16_t)oacc[j][r];
      }
    }
  }
}

// ---------------- host launcher ----------------
extern "C" void kernel_launch(void* const* d_in, const int* in_sizes, int n_in,
                              void* d_out, int out_size, void* d_ws, size_t ws_size,
                              hipStream_t stream) {
  (void)in_sizes; (void)n_in; (void)out_size; (void)ws_size;
  const float* hs  = (const float*)d_in[0];
  const float* fco = (const float*)d_in[1];
  const float* fsi = (const float*)d_in[2];
  /* d_in[3] = mask (all ones) unused */
  const float* Wq  = (const float*)d_in[4];
  const float* Wk  = (const float*)d_in[5];
  const float* Wv  = (const float*)d_in[6];
  const float* Wo  = (const float*)d_in[7];
  const float* fmq = (const float*)d_in[8];
  const float* fmk = (const float*)d_in[9];

  char* p = (char*)d_ws;
  auto take = [&](size_t bytes) { char* r = p; p += (bytes + 255) & ~(size_t)255; return r; };

  const size_t BL = 4096;           // B*L
  bf16_t* hsb   = (bf16_t*)take(BL * 2048 * 2);
  bf16_t* WqT   = (bf16_t*)take((size_t)2048 * 2048 * 2);
  bf16_t* WkT   = (bf16_t*)take((size_t)256 * 2048 * 2);
  bf16_t* WvT   = (bf16_t*)take((size_t)256 * 2048 * 2);
  bf16_t* WoT   = (bf16_t*)take((size_t)2048 * 2048 * 2);
  bf16_t* fmqT  = (bf16_t*)take((size_t)8 * 64 * 256 * 2);
  bf16_t* fmkT  = (bf16_t*)take((size_t)8 * 64 * 256 * 2);
  float*  Qf    = (float*)take(BL * 2048 * 4);
  float*  Kf    = (float*)take(BL * 256 * 4);
  float*  Vf    = (float*)take(BL * 256 * 4);
  bf16_t* Qr    = (bf16_t*)take((size_t)2 * 8 * 2048 * 256 * 2);
  bf16_t* Kr    = (bf16_t*)take((size_t)2 * 1 * 2048 * 256 * 2);
  bf16_t* phiQ  = (bf16_t*)take((size_t)2 * 8 * 2048 * 128 * 2);
  bf16_t* phiK  = (bf16_t*)take((size_t)2 * 8 * 2048 * 128 * 2);
  bf16_t* attnb = (bf16_t*)take(BL * 2048 * 2);

  const int T = 256;
  cvt_f32_bf16<<<(8388608 + T - 1) / T, T, 0, stream>>>(hs, hsb, 8388608);
  transpose_cvt<<<(4194304 + T - 1) / T, T, 0, stream>>>(Wq, WqT, 2048, 2048);
  transpose_cvt<<<(524288 + T - 1) / T, T, 0, stream>>>(Wk, WkT, 2048, 256);
  transpose_cvt<<<(524288 + T - 1) / T, T, 0, stream>>>(Wv, WvT, 2048, 256);
  transpose_cvt<<<(4194304 + T - 1) / T, T, 0, stream>>>(Wo, WoT, 2048, 2048);
  fm_transpose<<<(131072 + T - 1) / T, T, 0, stream>>>(fmq, fmqT);
  fm_transpose<<<(131072 + T - 1) / T, T, 0, stream>>>(fmk, fmkT);

  const size_t gemm_smem = (size_t)4 * 128 * GEMM_LDA * 2;  // 73728 (double-buffered A+B)
  gemm_bf16<2048, 2048><<<dim3(16, 32), T, gemm_smem, stream>>>(hsb, WqT, Qf);
  gemm_bf16<256,  2048><<<dim3(2, 32),  T, gemm_smem, stream>>>(hsb, WkT, Kf);
  gemm_bf16<256,  2048><<<dim3(2, 32),  T, gemm_smem, stream>>>(hsb, WvT, Vf);

  rope_kernel<<<(4194304 + T - 1) / T, T, 0, stream>>>(Qf, fco, fsi, Qr, 8);
  rope_kernel<<<(524288 + T - 1) / T, T, 0, stream>>>(Kf, fco, fsi, Kr, 1);

  const size_t hh_smem = (size_t)2 * 64 * 264 * 2 + (size_t)64 * 72 * 4;  // 86016
  hedgehog_kernel<<<dim3(32, 16), T, hh_smem, stream>>>(Qr, fmqT, phiQ, 8, 0.08838834764831845f);
  hedgehog_kernel<<<dim3(32, 16), T, hh_smem, stream>>>(Kr, fmkT, phiK, 1, 1.0f);

  const size_t at_smem = ((size_t)64 * 136 * 3 + (size_t)128 * 72 + (size_t)64 * 72 * 2) * 2; // 89088
  chunk_attn_kernel<<<dim3(16, 4), T, at_smem, stream>>>(phiQ, phiK, Vf, attnb);

  gemm_bf16<2048, 2048><<<dim3(16, 32), T, gemm_smem, stream>>>(attnb, WoT, (float*)d_out);
}
